// RGATBlock_72834055406011
// MI455X (gfx1250) — matrix operand order repaired and checked by reading, not compiled, not run on hardware
//
#include <hip/hip_runtime.h>

typedef __attribute__((ext_vector_type(16))) __bf16 v16bf;
typedef __attribute__((ext_vector_type(8)))  __bf16 v8bf;
typedef __attribute__((ext_vector_type(4)))  __bf16 v4bf;
typedef __attribute__((ext_vector_type(8)))  float  v8f;

#define HID 128
#define NHEAD 4

// ---------------------------------------------------------------------------
// Transpose f32 [rows][cols] -> bf16 [cols][rows], batched over blockIdx.z
// ---------------------------------------------------------------------------
__global__ __launch_bounds__(256) void k_transpose_bf16(
    const float* __restrict__ src, __bf16* __restrict__ dst, int rows, int cols) {
  long total = (long)rows * cols;
  long t = (long)blockIdx.x * blockDim.x + threadIdx.x;
  long boff = (long)blockIdx.z * total;
  if (t < total) {
    int r = (int)(t / cols), c = (int)(t % cols);
    dst[boff + (long)c * rows + r] = (__bf16)src[boff + t];
  }
}

// ---------------------------------------------------------------------------
// Per-relation edge-embedding attention coefficient: ae[r][h] = emb[r]·att_edge[h]
// ---------------------------------------------------------------------------
__global__ void k_ae(const float* __restrict__ emb, const float* __restrict__ att_edge,
                     float* __restrict__ ae, int R) {
  int t = blockIdx.x * blockDim.x + threadIdx.x;
  if (t < R * NHEAD) {
    int r = t >> 2, hh = t & 3;
    float s = 0.f;
    for (int d = 0; d < 32; ++d) s += emb[r * HID + hh * 32 + d] * att_edge[hh * 32 + d];
    ae[t] = s;
  }
}

// ---------------------------------------------------------------------------
// LayerNorm (wave per row, 128 cols) -> bf16
// ---------------------------------------------------------------------------
__global__ __launch_bounds__(256) void k_layernorm_bf16(
    const float* __restrict__ x, const float* __restrict__ w, const float* __restrict__ b,
    __bf16* __restrict__ out, long N) {
  int wave = threadIdx.x >> 5, lane = threadIdx.x & 31;
  long row = (long)blockIdx.x * 8 + wave;
  if (row >= N) return;
  float4 v = ((const float4*)(x + row * HID))[lane];
  float s  = v.x + v.y + v.z + v.w;
  float ss = v.x * v.x + v.y * v.y + v.z * v.z + v.w * v.w;
#pragma unroll
  for (int m = 16; m >= 1; m >>= 1) { s += __shfl_xor(s, m, 32); ss += __shfl_xor(ss, m, 32); }
  float mean = s * (1.f / HID);
  float rstd = rsqrtf(ss * (1.f / HID) - mean * mean + 1e-5f);
  float4 wv = ((const float4*)w)[lane];
  float4 bv = ((const float4*)b)[lane];
  v4bf o;
  o[0] = (__bf16)((v.x - mean) * rstd * wv.x + bv.x);
  o[1] = (__bf16)((v.y - mean) * rstd * wv.y + bv.y);
  o[2] = (__bf16)((v.z - mean) * rstd * wv.z + bv.z);
  o[3] = (__bf16)((v.w - mean) * rstd * wv.w + bv.w);
  *(v4bf*)(out + row * HID + lane * 4) = o;
}

// ---------------------------------------------------------------------------
// Generic WMMA GEMM: C[n][o] = A_bf16[n][K] @ Wt_bf16[o][K]^T  (+ epilogue)
// A and (for EPI 0/1/2) the output are row-padded to a multiple of 16 rows,
// so loads AND stores are unguarded; only EPI 3 (d_out) guards stores.
// 256 threads = 8 waves; wave -> one 16x16 output tile. blockIdx.z = batch.
// EPI: 0 = f32 out (+bias)           (root -> agg)
//      1 = bf16 out                  (relation projection -> xr)
//      2 = bf16 out, gelu(acc+bias)  (FFN1 -> f1)
//      3 = f32 out, resid+acc+bias   (FFN2 -> d_out)
// ---------------------------------------------------------------------------
template <int EPI, int K>
__global__ __launch_bounds__(256) void k_gemm_wmma(
    const __bf16* __restrict__ A, const __bf16* __restrict__ Wt,
    const float* __restrict__ bias, const float* __restrict__ resid,
    void* __restrict__ outv, int nRows, int nCols,
    long aBatch, long wBatch, long oBatch) {
  int wave = threadIdx.x >> 5, lane = threadIdx.x & 31;
  int row0 = blockIdx.x * 16;
  int col0 = (blockIdx.y * 8 + wave) * 16;
  if (col0 >= nCols) return;
  A  += (long)blockIdx.z * aBatch;
  Wt += (long)blockIdx.z * wBatch;

  int M = lane & 15, half = lane >> 4;
  // Per-lane fragment base pointers (contiguous per ISA layout tables):
  //   A frag: K in [k0+8h, k0+8h+8) and [k0+16+8h, k0+16+8h+8)  (lane = M row)
  //   B frag: K in [k0+8h, k0+8h+8) and [k0+16+8h, k0+16+8h+8)  (lane = N col)
  const __bf16* ap = A  + (long)(row0 + M) * K + half * 8;
  const __bf16* bp = Wt + (long)(col0 + M) * K + half * 8;

  v8f acc = {};
#pragma unroll
  for (int k0 = 0; k0 < K; k0 += 32) {
    v8bf a0 = *(const v8bf*)(ap + k0);
    v8bf a1 = *(const v8bf*)(ap + k0 + 16);
    v8bf b0 = *(const v8bf*)(bp + k0);
    v8bf b1 = *(const v8bf*)(bp + k0 + 16);
    v16bf a = __builtin_shufflevector(a0, a1, 0, 1, 2, 3, 4, 5, 6, 7,
                                      8, 9, 10, 11, 12, 13, 14, 15);
    v16bf b = __builtin_shufflevector(b0, b1, 0, 1, 2, 3, 4, 5, 6, 7,
                                      8, 9, 10, 11, 12, 13, 14, 15);
    acc = __builtin_amdgcn_wmma_f32_16x16x32_bf16(false, a, false, b, (short)0, acc,
                                                  false, false);
  }

  int ccol = col0 + (lane & 15);
  float bval = (bias != nullptr) ? bias[ccol] : 0.f;
#pragma unroll
  for (int v = 0; v < 8; ++v) {
    int crow = row0 + v + 8 * half;
    float val = acc[v] + bval;
    long idx = (long)crow * nCols + ccol;
    if constexpr (EPI == 0) {
      ((float*)outv + (long)blockIdx.z * oBatch)[idx] = val;
    } else if constexpr (EPI == 1) {
      ((__bf16*)outv + (long)blockIdx.z * oBatch)[idx] = (__bf16)val;
    } else if constexpr (EPI == 2) {
      float g = 0.5f * val * (1.0f + erff(val * 0.7071067811865475f));
      ((__bf16*)outv)[idx] = (__bf16)g;
    } else {  // EPI == 3: exact-size output, guard rows
      if (crow < nRows) ((float*)outv)[idx] = resid[idx] + val;
    }
  }
}

// ---------------------------------------------------------------------------
// Per (relation,node) attention dots: a_src/a_dst[rn][h] = xr[rn]·att_{src,dst}[h]
// wave per row; 8 lanes per head (4 elems each). Runs over R*Npad rows.
// ---------------------------------------------------------------------------
__global__ __launch_bounds__(256) void k_att(
    const __bf16* __restrict__ xr, const float* __restrict__ att_src,
    const float* __restrict__ att_dst, float* __restrict__ a_src,
    float* __restrict__ a_dst, long RN) {
  int wave = threadIdx.x >> 5, lane = threadIdx.x & 31;
  long row = (long)blockIdx.x * 8 + wave;
  if (row >= RN) return;
  v4bf xv = *(const v4bf*)(xr + row * HID + lane * 4);
  float x0 = (float)xv[0], x1 = (float)xv[1], x2 = (float)xv[2], x3 = (float)xv[3];
  int base = lane * 4;
  float s1 = x0 * att_src[base] + x1 * att_src[base + 1] + x2 * att_src[base + 2] + x3 * att_src[base + 3];
  float s2 = x0 * att_dst[base] + x1 * att_dst[base + 1] + x2 * att_dst[base + 2] + x3 * att_dst[base + 3];
#pragma unroll
  for (int m = 1; m <= 4; m <<= 1) { s1 += __shfl_xor(s1, m, 32); s2 += __shfl_xor(s2, m, 32); }
  if ((lane & 7) == 0) {
    a_src[row * NHEAD + (lane >> 3)] = s1;
    a_dst[row * NHEAD + (lane >> 3)] = s2;
  }
}

__device__ __forceinline__ unsigned fmap(float f) {
  unsigned u = __float_as_uint(f);
  return (u >> 31) ? ~u : (u | 0x80000000u);
}
__device__ __forceinline__ float funmap(unsigned u) {
  return (u >> 31) ? __uint_as_float(u ^ 0x80000000u) : __uint_as_float(~u);
}

// ---------------------------------------------------------------------------
// Edge phase 1: raw alpha = leaky(a_src+a_dst+ae); atomic segment max over dst
// thread per (edge, head). NP = padded per-relation node stride.
// ---------------------------------------------------------------------------
__global__ __launch_bounds__(256) void k_edge_alpha(
    const int* __restrict__ ei, const int* __restrict__ et,
    const float* __restrict__ a_src, const float* __restrict__ a_dst,
    const float* __restrict__ ae, float* __restrict__ alpha,
    unsigned* __restrict__ segmax, long NP, long E) {
  long t = (long)blockIdx.x * blockDim.x + threadIdx.x;
  if (t >= E * NHEAD) return;
  long e = t >> 2; int hh = (int)(t & 3);
  int r = et[e], s = ei[e], d = ei[E + e];
  float a = a_src[((long)r * NP + s) * NHEAD + hh] + a_dst[((long)r * NP + d) * NHEAD + hh] +
            ae[r * NHEAD + hh];
  a = a > 0.f ? a : 0.2f * a;
  alpha[t] = a;
  atomicMax(&segmax[(long)d * NHEAD + hh], fmap(a));
}

// Edge phase 2: ex = exp(alpha - max[dst]); atomic segment sum
__global__ __launch_bounds__(256) void k_edge_exp(
    const int* __restrict__ ei, float* __restrict__ alpha,
    const unsigned* __restrict__ segmax, float* __restrict__ segsum, long E) {
  long t = (long)blockIdx.x * blockDim.x + threadIdx.x;
  if (t >= E * NHEAD) return;
  long e = t >> 2; int hh = (int)(t & 3);
  int d = ei[E + e];
  float m  = funmap(segmax[(long)d * NHEAD + hh]);
  float ex = __expf(alpha[t] - m);
  alpha[t] = ex;
  atomicAdd(&segsum[(long)d * NHEAD + hh], ex);
}

// Edge phase 3: messages — wave per edge, scatter-add x_src * alpha_norm
__global__ __launch_bounds__(256) void k_edge_msg(
    const int* __restrict__ ei, const int* __restrict__ et,
    const __bf16* __restrict__ xr, const float* __restrict__ alpha,
    const float* __restrict__ segsum, float* __restrict__ agg, long NP, long E) {
  int wave = threadIdx.x >> 5, lane = threadIdx.x & 31;
  long e = (long)blockIdx.x * 8 + wave;
  if (e >= E) return;
  int r = et[e], s = ei[e], d = ei[E + e];
  int hh = lane >> 3;
  float ex  = alpha[e * NHEAD + hh];
  float den = segsum[(long)d * NHEAD + hh];
  float an  = ex / (den + 1e-16f);
  v4bf xv = *(const v4bf*)(xr + ((long)r * NP + s) * HID + lane * 4);
  float* ap = agg + (long)d * HID + lane * 4;
  atomicAdd(ap + 0, (float)xv[0] * an);
  atomicAdd(ap + 1, (float)xv[1] * an);
  atomicAdd(ap + 2, (float)xv[2] * an);
  atomicAdd(ap + 3, (float)xv[3] * an);
}

// ---------------------------------------------------------------------------
// Fused: x2 = x + elu(agg); h2b = bf16(LayerNorm(x2))  (wave per row)
// ---------------------------------------------------------------------------
__global__ __launch_bounds__(256) void k_fuse_mid(
    const float* __restrict__ x, const float* __restrict__ agg,
    const float* __restrict__ w, const float* __restrict__ b,
    float* __restrict__ x2, __bf16* __restrict__ h2b, long N) {
  int wave = threadIdx.x >> 5, lane = threadIdx.x & 31;
  long row = (long)blockIdx.x * 8 + wave;
  if (row >= N) return;
  float4 xv = ((const float4*)(x + row * HID))[lane];
  float4 av = ((const float4*)(agg + row * HID))[lane];
  float t0 = xv.x + (av.x > 0.f ? av.x : __expf(av.x) - 1.f);
  float t1 = xv.y + (av.y > 0.f ? av.y : __expf(av.y) - 1.f);
  float t2 = xv.z + (av.z > 0.f ? av.z : __expf(av.z) - 1.f);
  float t3 = xv.w + (av.w > 0.f ? av.w : __expf(av.w) - 1.f);
  float s  = t0 + t1 + t2 + t3;
  float ss = t0 * t0 + t1 * t1 + t2 * t2 + t3 * t3;
#pragma unroll
  for (int m = 16; m >= 1; m >>= 1) { s += __shfl_xor(s, m, 32); ss += __shfl_xor(ss, m, 32); }
  float mean = s * (1.f / HID);
  float rstd = rsqrtf(ss * (1.f / HID) - mean * mean + 1e-5f);
  float4 o; o.x = t0; o.y = t1; o.z = t2; o.w = t3;
  ((float4*)(x2 + row * HID))[lane] = o;
  float4 wv = ((const float4*)w)[lane];
  float4 bv = ((const float4*)b)[lane];
  v4bf hv;
  hv[0] = (__bf16)((t0 - mean) * rstd * wv.x + bv.x);
  hv[1] = (__bf16)((t1 - mean) * rstd * wv.y + bv.y);
  hv[2] = (__bf16)((t2 - mean) * rstd * wv.z + bv.z);
  hv[3] = (__bf16)((t3 - mean) * rstd * wv.w + bv.w);
  *(v4bf*)(h2b + row * HID + lane * 4) = hv;
}

// ---------------------------------------------------------------------------
extern "C" void kernel_launch(void* const* d_in, const int* in_sizes, int n_in,
                              void* d_out, int out_size, void* d_ws, size_t ws_size,
                              hipStream_t stream) {
  (void)n_in; (void)out_size; (void)ws_size;
  const long N    = in_sizes[0] / HID;
  const long Npad = (N + 15) & ~15L;
  const int  R    = in_sizes[1] / (HID * HID);
  const int  FF   = in_sizes[13];
  const long E    = in_sizes[17];

  const float* x        = (const float*)d_in[0];
  const float* weight   = (const float*)d_in[1];
  const float* root_w   = (const float*)d_in[2];
  const float* edge_emb = (const float*)d_in[3];
  const float* att_src  = (const float*)d_in[4];
  const float* att_dst  = (const float*)d_in[5];
  const float* att_edge = (const float*)d_in[6];
  const float* bias     = (const float*)d_in[7];
  const float* n1w      = (const float*)d_in[8];
  const float* n1b      = (const float*)d_in[9];
  const float* n2w      = (const float*)d_in[10];
  const float* n2b      = (const float*)d_in[11];
  const float* ffn_w1   = (const float*)d_in[12];
  const float* ffn_b1   = (const float*)d_in[13];
  const float* ffn_w2   = (const float*)d_in[14];
  const float* ffn_b2   = (const float*)d_in[15];
  const int*   ei       = (const int*)d_in[16];
  const int*   et       = (const int*)d_in[17];

  // workspace carve (256B aligned)
  char* base = (char*)d_ws; size_t off = 0;
  auto carve = [&](size_t bytes) -> char* {
    char* p = base + off;
    off += (bytes + 255) & ~(size_t)255;
    return p;
  };
  __bf16*   wtb    = (__bf16*)carve((size_t)R * HID * HID * 2);     // [R][out][in]
  __bf16*   rootb  = (__bf16*)carve((size_t)HID * HID * 2);         // [out][in]
  __bf16*   w1t    = (__bf16*)carve((size_t)FF * HID * 2);          // [FF][HID]
  __bf16*   w2t    = (__bf16*)carve((size_t)HID * FF * 2);          // [HID][FF]
  float*    ae     = (float*)carve((size_t)R * NHEAD * 4);
  __bf16*   hb     = (__bf16*)carve((size_t)Npad * HID * 2);        // LN1 out (padded)
  __bf16*   xrb    = (__bf16*)carve((size_t)R * Npad * HID * 2);    // projections (padded)
  float*    a_src  = (float*)carve((size_t)R * Npad * NHEAD * 4);
  float*    a_dst  = (float*)carve((size_t)R * Npad * NHEAD * 4);
  float*    alpha  = (float*)carve((size_t)E * NHEAD * 4);
  unsigned* segmax = (unsigned*)carve((size_t)N * NHEAD * 4);
  float*    segsum = (float*)carve((size_t)N * NHEAD * 4);
  float*    agg    = (float*)carve((size_t)Npad * HID * 4);         // root+bias+msgs (padded)
  float*    x2     = (float*)carve((size_t)N * HID * 4);            // mid residual
  __bf16*   h2b    = (__bf16*)carve((size_t)Npad * HID * 2);        // LN2 out (padded)
  __bf16*   f1b    = (__bf16*)carve((size_t)Npad * FF * 2);         // gelu(ffn1) (padded)

  dim3 blk(256);
  int rowTiles = (int)(Npad / 16);

  // zero-fill pad tails so GEMM A-loads need no guards
  if (Npad > N) {
    hipMemsetAsync(hb  + N * HID, 0, (size_t)(Npad - N) * HID * 2, stream);
    hipMemsetAsync(h2b + N * HID, 0, (size_t)(Npad - N) * HID * 2, stream);
  }

  // weight conversion + transpose to bf16 [out][in]
  k_transpose_bf16<<<dim3((HID * HID + 255) / 256, 1, R), blk, 0, stream>>>(weight, wtb, HID, HID);
  k_transpose_bf16<<<dim3((HID * HID + 255) / 256, 1, 1), blk, 0, stream>>>(root_w, rootb, HID, HID);
  k_transpose_bf16<<<dim3((HID * FF + 255) / 256, 1, 1), blk, 0, stream>>>(ffn_w1, w1t, HID, FF);
  k_transpose_bf16<<<dim3((HID * FF + 255) / 256, 1, 1), blk, 0, stream>>>(ffn_w2, w2t, FF, HID);
  k_ae<<<1, 256, 0, stream>>>(edge_emb, att_edge, ae, R);

  // LN1 -> bf16
  k_layernorm_bf16<<<(unsigned)((N + 7) / 8), blk, 0, stream>>>(x, n1w, n1b, hb, N);

  // relation projections (bf16 WMMA, bf16 out) and root GEMM (f32 out + bias)
  k_gemm_wmma<1, HID><<<dim3(rowTiles, 1, R), blk, 0, stream>>>(
      hb, wtb, nullptr, nullptr, (void*)xrb, (int)Npad, HID,
      0L, (long)HID * HID, (long)Npad * HID);
  k_gemm_wmma<0, HID><<<dim3(rowTiles, 1, 1), blk, 0, stream>>>(
      hb, rootb, bias, nullptr, (void*)agg, (int)Npad, HID, 0L, 0L, 0L);

  // attention coefficients per (relation, padded node)
  k_att<<<(unsigned)(((long)R * Npad + 7) / 8), blk, 0, stream>>>(
      xrb, att_src, att_dst, a_src, a_dst, (long)R * Npad);

  hipMemsetAsync(segmax, 0, (size_t)N * NHEAD * 4, stream);
  hipMemsetAsync(segsum, 0, (size_t)N * NHEAD * 4, stream);

  unsigned ehBlocks = (unsigned)((E * NHEAD + 255) / 256);
  k_edge_alpha<<<ehBlocks, blk, 0, stream>>>(ei, et, a_src, a_dst, ae, alpha, segmax, Npad, E);
  k_edge_exp<<<ehBlocks, blk, 0, stream>>>(ei, alpha, segmax, segsum, E);
  k_edge_msg<<<(unsigned)((E + 7) / 8), blk, 0, stream>>>(ei, et, xrb, alpha, segsum, agg, Npad, E);

  // x2 = x + elu(conv); h2b = LN2(x2) in bf16
  k_fuse_mid<<<(unsigned)((N + 7) / 8), blk, 0, stream>>>(x, agg, n2w, n2b, x2, h2b, N);

  // FFN: gelu(h2 @ w1 + b1) -> bf16 ; out = x2 + f1 @ w2 + b2
  k_gemm_wmma<2, HID><<<dim3(rowTiles, FF / 128, 1), blk, 0, stream>>>(
      h2b, w1t, ffn_b1, nullptr, (void*)f1b, (int)Npad, FF, 0L, 0L, 0L);
  k_gemm_wmma<3, 4 * HID><<<dim3(rowTiles, 1, 1), blk, 0, stream>>>(
      f1b, w2t, ffn_b2, x2, d_out, (int)N, HID, 0L, 0L, 0L);
}
